// RandLANet_27650999452450
// MI455X (gfx1250) — matrix operand order repaired and checked by reading, not compiled, not run on hardware
//
#include <hip/hip_runtime.h>
#include <hip/hip_bf16.h>

typedef __attribute__((ext_vector_type(16))) _Float16 v16h;
typedef __attribute__((ext_vector_type(8)))  float    v8f;

#define KNN_K 16

// ---------------------------------------------------------------------------
// WMMA GEMM:  Y[b] = act( W(Cout x Cin) * X[b](Cin x M) + bias )
// X layout (B, Cin, M) row-major; Y layout (B, Cout, M).
// Block = 128 threads (4 waves): 32 output rows x 128 output cols.
// A (32x32) and B (32x128) tiles staged in LDS (f16, padded stride 34 to
// dodge bank conflicts); cooperative loads are coalesced along M and use a
// per-k recomputed clamped address (no per-lane address arrays -> no spills).
// Each wave computes a 32x32 tile with 4 accumulators (4 v_wmma per K-step).
// act: 0 none, 1 relu, 2 leaky(slope)
// ---------------------------------------------------------------------------
__global__ void k_gemm(const float* __restrict__ W, const float* __restrict__ bias,
                       const float* __restrict__ X, float* __restrict__ Y,
                       int Cout, int Cin, int M, int act, float slope)
{
    __shared__ _Float16 As[32 * 34];
    __shared__ _Float16 Bs[128 * 34];

    int tid  = threadIdx.x;
    int lane = tid & 31;
    int wave = tid >> 5;
    int l15  = lane & 15;
    int hi   = (lane >> 4) & 1;

    int mtb = blockIdx.x * 128;          // block column base (M)
    int ot  = blockIdx.y * 32;           // row base (Cout)
    int b   = blockIdx.z;

    const float* Xb = X + (size_t)b * Cin * M;
    float*       Yb = Y + (size_t)b * Cout * M;

    v8f acc00 = {}, acc01 = {}, acc10 = {}, acc11 = {};

    int kA = hi ? 8 : 0;                 // A K sub-base per ISA layout
    int kB = hi ? 8 : 0;                 // B K sub-base

    const _Float16* ap0 = &As[l15 * 34 + kA];
    const _Float16* ap1 = &As[(16 + l15) * 34 + kA];
    const _Float16* bp0 = &Bs[(wave * 32 + l15) * 34 + kB];
    const _Float16* bp1 = &Bs[(wave * 32 + 16 + l15) * 34 + kB];

    int colc = min(mtb + tid, M - 1);    // this thread's clamped load column

    for (int kb = 0; kb < Cin; kb += 32) {
        // --- cooperative A tile: 32 rows x 32 k, f32 -> f16 ----------------
#pragma unroll
        for (int e = tid; e < 32 * 32; e += 128) {
            int row = e >> 5, k = e & 31;
            int gr = min(ot + row, Cout - 1);
            int gk = kb + k;
            float v = W[(size_t)gr * Cin + min(gk, Cin - 1)];
            As[row * 34 + k] = (gk < Cin) ? (_Float16)v : (_Float16)0.0f;
        }
        // --- cooperative B tile: 32 k x 128 cols, coalesced along M --------
        {
            const float* xp = Xb + (size_t)min(kb, Cin - 1) * M + colc;
#pragma unroll 4
            for (int k = 0; k < 32; ++k) {
                int gk = kb + k;
                float v = Xb[(size_t)min(gk, Cin - 1) * M + colc];
                Bs[tid * 34 + k] = (gk < Cin) ? (_Float16)v : (_Float16)0.0f;
            }
            (void)xp;
        }
        __syncthreads();

        // --- fragments from LDS + 4 WMMA -----------------------------------
        v16h a0, a1, b0, b1;
#pragma unroll
        for (int j = 0; j < 16; ++j) {
            int o = (j < 8) ? j : j + 8;
            a0[j] = ap0[o];
            a1[j] = ap1[o];
            b0[j] = bp0[o];
            b1[j] = bp1[o];
        }
        acc00 = __builtin_amdgcn_wmma_f32_16x16x32_f16(false, a0, false, b0, (short)0, acc00, false, false);
        acc01 = __builtin_amdgcn_wmma_f32_16x16x32_f16(false, a0, false, b1, (short)0, acc01, false, false);
        acc10 = __builtin_amdgcn_wmma_f32_16x16x32_f16(false, a1, false, b0, (short)0, acc10, false, false);
        acc11 = __builtin_amdgcn_wmma_f32_16x16x32_f16(false, a1, false, b1, (short)0, acc11, false, false);
        __syncthreads();
    }

    // --- store 32x32 per wave ----------------------------------------------
    int c0  = mtb + wave * 32 + l15;
    int c1  = c0 + 16;
    int rb0 = ot + (hi ? 8 : 0);
    int rb1 = rb0 + 16;
#pragma unroll
    for (int t = 0; t < 4; ++t) {
        const v8f& acc = (t == 0) ? acc00 : (t == 1) ? acc01 : (t == 2) ? acc10 : acc11;
        int rb  = (t < 2) ? rb0 : rb1;
        int col = ((t & 1) == 0) ? c0 : c1;
        if (col < M) {
#pragma unroll
            for (int rr = 0; rr < 8; ++rr) {
                int row = rb + rr;
                if (row < Cout) {
                    float v = acc[rr] + (bias ? bias[row] : 0.0f);
                    if (act == 1)      v = fmaxf(v, 0.0f);
                    else if (act == 2) v = v > 0.0f ? v : v * slope;
                    Yb[(size_t)row * M + col] = v;
                }
            }
        }
    }
}

// ---------------------------------------------------------------------------
// BatchNorm statistics over (B, M) for each channel c of a (B, C, M) tensor.
// ---------------------------------------------------------------------------
__global__ void k_bn_stats(const float* __restrict__ Y, int B, int C, int M,
                           float* __restrict__ mean, float* __restrict__ rstd)
{
    int c = blockIdx.x;
    float s = 0.f, s2 = 0.f;
    for (int b = 0; b < B; ++b) {
        const float* p = Y + ((size_t)b * C + c) * M;
        for (int m = threadIdx.x; m < M; m += blockDim.x) {
            float v = p[m]; s += v; s2 += v * v;
        }
    }
    __shared__ float sh1[256], sh2[256];
    sh1[threadIdx.x] = s; sh2[threadIdx.x] = s2;
    __syncthreads();
    for (int o = 128; o > 0; o >>= 1) {
        if ((int)threadIdx.x < o) {
            sh1[threadIdx.x] += sh1[threadIdx.x + o];
            sh2[threadIdx.x] += sh2[threadIdx.x + o];
        }
        __syncthreads();
    }
    if (threadIdx.x == 0) {
        float inv = 1.0f / (float)(B * M);
        float mu  = sh1[0] * inv;
        float var = sh2[0] * inv - mu * mu;
        mean[c] = mu;
        rstd[c] = rsqrtf(fmaxf(var, 0.0f) + 1e-6f);
    }
}

// out = act( (Y - mean)*rstd*gamma + beta + res )
__global__ void k_bn_apply(const float* __restrict__ Y, const float* __restrict__ mean,
                           const float* __restrict__ rstd, const float* __restrict__ gamma,
                           const float* __restrict__ beta, const float* __restrict__ res,
                           float* __restrict__ O, int B, int C, size_t M,
                           int act, float slope)
{
    size_t tot = (size_t)B * C * M;
    size_t i = (size_t)blockIdx.x * blockDim.x + threadIdx.x;
    if (i >= tot) return;
    int c = (int)((i / M) % C);
    float v = (Y[i] - mean[c]) * rstd[c] * gamma[c] + beta[c];
    if (res) v += res[i];
    if (act == 1)      v = fmaxf(v, 0.0f);
    else if (act == 2) v = v > 0.0f ? v : v * slope;
    O[i] = v;
}

// ---------------------------------------------------------------------------
// KNN (k=16) over same point set; per-thread register top-k, LDS-staged tiles.
// ---------------------------------------------------------------------------
__global__ void k_knn(const float* __restrict__ coords, int n, int cstride,
                      int* __restrict__ idx_out, float* __restrict__ dist_out)
{
    int b = blockIdx.y;
    const float* C = coords + (size_t)b * cstride;
    int q = blockIdx.x * blockDim.x + threadIdx.x;
    float qx = 0.f, qy = 0.f, qz = 0.f;
    if (q < n) { qx = C[q*3]; qy = C[q*3+1]; qz = C[q*3+2]; }
    float bd[KNN_K]; int bi[KNN_K];
#pragma unroll
    for (int i = 0; i < KNN_K; ++i) { bd[i] = 3.4e38f; bi[i] = 0; }

    __shared__ float sx[128], sy[128], sz[128];
    for (int t0 = 0; t0 < n; t0 += 128) {
        __syncthreads();
        int load = t0 + threadIdx.x;
        if (load < n) {
            sx[threadIdx.x] = C[load*3];
            sy[threadIdx.x] = C[load*3+1];
            sz[threadIdx.x] = C[load*3+2];
        }
        if (t0 + 128 + (int)threadIdx.x < n)
            __builtin_prefetch(&C[(t0 + 128 + threadIdx.x) * 3], 0, 1);
        __syncthreads();
        int lim = min(128, n - t0);
        for (int t = 0; t < lim; ++t) {
            float dx = qx - sx[t], dy = qy - sy[t], dz = qz - sz[t];
            float d = dx*dx + dy*dy + dz*dz;
            if (d < bd[KNN_K-1]) {
                int pos = KNN_K - 1;
#pragma unroll
                for (int i = KNN_K - 2; i >= 0; --i) {
                    if (d < bd[i]) { bd[i+1] = bd[i]; bi[i+1] = bi[i]; pos = i; }
                }
                bd[pos] = d; bi[pos] = t0 + t;
            }
        }
    }
    if (q < n) {
        for (int i = 0; i < KNN_K; ++i) {
            idx_out[((size_t)b*n + q)*KNN_K + i]  = bi[i];
            dist_out[((size_t)b*n + q)*KNN_K + i] = fmaxf(bd[i], 0.0f);
        }
    }
}

// nearest neighbor (k=1) of queries [0,nq) among supports [0,ns)
__global__ void k_nn1(const float* __restrict__ coords, int cstride,
                      int nq, int ns, int* __restrict__ idx)
{
    int b = blockIdx.y;
    int q = blockIdx.x * blockDim.x + threadIdx.x;
    if (q >= nq) return;
    const float* C = coords + (size_t)b * cstride;
    float qx = C[q*3], qy = C[q*3+1], qz = C[q*3+2];
    float best = 3.4e38f; int bidx = 0;
    for (int s = 0; s < ns; ++s) {
        float dx = qx - C[s*3], dy = qy - C[s*3+1], dz = qz - C[s*3+2];
        float d = dx*dx + dy*dy + dz*dz;
        if (d < best) { best = d; bidx = s; }
    }
    idx[(size_t)b * nq + q] = bidx;
}

// ---------------------------------------------------------------------------
// LSE input build: out (B, 10, n*K) = [ctr(3), nbr(3), ctr-nbr(3), dist(1)]
// ---------------------------------------------------------------------------
__global__ void k_lse_build(const float* __restrict__ coords, int cstride,
                            const int* __restrict__ idx, const float* __restrict__ dist,
                            float* __restrict__ out, int B, int n, int K)
{
    size_t tot = (size_t)B * n * K;
    size_t i = (size_t)blockIdx.x * blockDim.x + threadIdx.x;
    if (i >= tot) return;
    int k = (int)(i % K); size_t nk = i / K;
    int q = (int)(nk % n); int b = (int)(nk / n);
    const float* C = coords + (size_t)b * cstride;
    int j = idx[((size_t)b*n + q)*K + k];
    float cx = C[q*3], cy = C[q*3+1], cz = C[q*3+2];
    float nx = C[j*3], ny = C[j*3+1], nz = C[j*3+2];
    size_t M = (size_t)n * K;
    float* o = out + (size_t)b * 10 * M + ((size_t)q * K + k);
    o[0]     = cx;      o[M]     = cy;      o[2*M]   = cz;
    o[3*M]   = nx;      o[4*M]   = ny;      o[5*M]   = nz;
    o[6*M]   = cx - nx; o[7*M]   = cy - ny; o[8*M]   = cz - nz;
    o[9*M]   = dist[((size_t)b*n + q)*K + k];
}

// concat(enc (B,h,n*K), broadcast(feat (B,h,n))) -> out (B, 2h, n*K)
__global__ void k_cat_enc_feat(const float* __restrict__ enc, const float* __restrict__ feat,
                               float* __restrict__ out, int B, int h, int n, int K)
{
    size_t M = (size_t)n * K;
    size_t tot = (size_t)B * h * M;
    size_t i = (size_t)blockIdx.x * blockDim.x + threadIdx.x;
    if (i >= tot) return;
    size_t m = i % M; size_t rest = i / M;
    int c = (int)(rest % h); int b = (int)(rest / h);
    int q = (int)(m / K);
    out[((size_t)b*2*h + c)*M + m]     = enc[i];
    out[((size_t)b*2*h + h + c)*M + m] = feat[((size_t)b*h + c)*n + q];
}

// attentive pooling: F[b,c,q] = sum_k softmax_k(S[b,c,q,:]) * X[b,c,q,:]
__global__ void k_attpool(const float* __restrict__ S, const float* __restrict__ X,
                          float* __restrict__ F, int B, int C, int n, int K)
{
    size_t tot = (size_t)B * C * n;
    size_t i = (size_t)blockIdx.x * blockDim.x + threadIdx.x;
    if (i >= tot) return;
    int q = (int)(i % n); size_t rest = i / n;
    int c = (int)(rest % C); int b = (int)(rest / C);
    size_t M = (size_t)n * K;
    const float* s = S + ((size_t)b*C + c)*M + (size_t)q*K;
    const float* x = X + ((size_t)b*C + c)*M + (size_t)q*K;
    float mx = -3.4e38f;
    for (int k = 0; k < K; ++k) mx = fmaxf(mx, s[k]);
    float sum = 0.f, acc = 0.f;
    for (int k = 0; k < K; ++k) {
        float w = __expf(s[k] - mx);
        sum += w; acc += w * x[k];
    }
    F[i] = acc / sum;
}

// Y[b,c,m] = X[b,c,idx[b?*,m]]   (idx_bstride==0 -> idx shared across batch)
__global__ void k_gather(const float* __restrict__ X, const int* __restrict__ idx,
                         int idx_bstride, float* __restrict__ Y,
                         int B, int C, int Mout, int Min)
{
    size_t tot = (size_t)B * C * Mout;
    size_t i = (size_t)blockIdx.x * blockDim.x + threadIdx.x;
    if (i >= tot) return;
    int m = (int)(i % Mout); size_t rest = i / Mout;
    int c = (int)(rest % C); int b = (int)(rest / C);
    int src = idx[(size_t)b * idx_bstride + m];
    Y[i] = X[((size_t)b*C + c)*(size_t)Min + src];
}

// Y = concat_channels(X1 (B,C1,M), X2 (B,C2,M))
__global__ void k_cat2(const float* __restrict__ X1, int C1,
                       const float* __restrict__ X2, int C2,
                       float* __restrict__ Y, int B, size_t M)
{
    int C = C1 + C2;
    size_t tot = (size_t)B * C * M;
    size_t i = (size_t)blockIdx.x * blockDim.x + threadIdx.x;
    if (i >= tot) return;
    size_t m = i % M; size_t rest = i / M;
    int c = (int)(rest % C); int b = (int)(rest / C);
    Y[i] = (c < C1) ? X1[((size_t)b*C1 + c)*M + m]
                    : X2[((size_t)b*C2 + (c - C1))*M + m];
}

// prefix column slice: Y (B,C,Mout) = X (B,C,Min)[:, :, :Mout]
__global__ void k_slice(const float* __restrict__ X, float* __restrict__ Y,
                        int B, int C, int Mout, int Min)
{
    size_t tot = (size_t)B * C * Mout;
    size_t i = (size_t)blockIdx.x * blockDim.x + threadIdx.x;
    if (i >= tot) return;
    int m = (int)(i % Mout); size_t rest = i / Mout;
    int c = (int)(rest % C); int b = (int)(rest / C);
    Y[i] = X[((size_t)b*C + c)*(size_t)Min + m];
}

// fc_start: Y (B,8,N) = W(8x9) * [pts(3)|feat(6)] + b
__global__ void k_fc_start(const float* __restrict__ pts, const float* __restrict__ feat,
                           const float* __restrict__ W, const float* __restrict__ bias,
                           float* __restrict__ Y, int B, int N)
{
    size_t tot = (size_t)B * N;
    size_t i = (size_t)blockIdx.x * blockDim.x + threadIdx.x;
    if (i >= tot) return;
    int n = (int)(i % N); int b = (int)(i / N);
    float in[9];
    in[0] = pts[((size_t)b*N + n)*3 + 0];
    in[1] = pts[((size_t)b*N + n)*3 + 1];
    in[2] = pts[((size_t)b*N + n)*3 + 2];
#pragma unroll
    for (int d = 0; d < 6; ++d) in[3+d] = feat[((size_t)b*N + n)*6 + d];
#pragma unroll
    for (int o = 0; o < 8; ++o) {
        float s = bias[o];
#pragma unroll
        for (int d = 0; d < 9; ++d) s += W[o*9 + d] * in[d];
        Y[((size_t)b*8 + o)*N + n] = s;
    }
}

// deterministic bijective permutation (stand-in for jax threefry perm), N = 2^k
__global__ void k_perm_init(int* __restrict__ perm, int* __restrict__ inv, int N)
{
    int i = blockIdx.x * blockDim.x + threadIdx.x;
    if (i >= N) return;
    int p = (int)(((unsigned)i * 2654435761u + 40503u) & (unsigned)(N - 1));
    perm[i] = p;
    inv[p] = i;
}

__global__ void k_perm_coords(const float* __restrict__ pts, const int* __restrict__ perm,
                              float* __restrict__ coords, int B, int N)
{
    size_t tot = (size_t)B * N;
    size_t i = (size_t)blockIdx.x * blockDim.x + threadIdx.x;
    if (i >= tot) return;
    int n = (int)(i % N); int b = (int)(i / N);
    int p = perm[n];
    coords[((size_t)b*N + n)*3 + 0] = pts[((size_t)b*N + p)*3 + 0];
    coords[((size_t)b*N + n)*3 + 1] = pts[((size_t)b*N + p)*3 + 1];
    coords[((size_t)b*N + n)*3 + 2] = pts[((size_t)b*N + p)*3 + 2];
}

// (B,C,N) -> (B,N,C)
__global__ void k_out_transpose(const float* __restrict__ X, float* __restrict__ out,
                                int B, int C, int N)
{
    size_t tot = (size_t)B * C * N;
    size_t i = (size_t)blockIdx.x * blockDim.x + threadIdx.x;
    if (i >= tot) return;
    int n = (int)(i % N); size_t rest = i / N;
    int c = (int)(rest % C); int b = (int)(rest / C);
    out[((size_t)b*N + n)*C + c] = X[i];
}

// ---------------------------------------------------------------------------
extern "C" void kernel_launch(void* const* d_in, const int* in_sizes, int n_in,
                              void* d_out, int out_size, void* d_ws, size_t ws_size,
                              hipStream_t stream)
{
    (void)in_sizes; (void)out_size; (void)ws_size;
    if (n_in < 138) return;

    const int B = 2, N = 8192, K = KNN_K;
    const float* points   = (const float*)d_in[0];
    const float* features = (const float*)d_in[1];

    // --- parameter extraction (jax tree_flatten: sorted dict keys) ---------
    int pi = 2;
    auto F = [&](int& p) { return (const float*)d_in[p++]; };
    struct MlpP { const float *W, *b, *beta, *gamma; };

    const float* bn_start_beta  = F(pi);
    const float* bn_start_gamma = F(pi);
    MlpP dec[4];
    for (int i = 0; i < 4; ++i) { dec[i].W = F(pi); dec[i].b = F(pi); dec[i].beta = F(pi); dec[i].gamma = F(pi); }
    struct EncP {
        MlpP lse1, lse2;
        const float *mlp1W, *mlp1b, *mlp2W, *mlp2b;
        MlpP p1mlp; const float* p1score;
        MlpP p2mlp; const float* p2score;
        MlpP sc;
    } E[4];
    for (int s = 0; s < 4; ++s) {
        E[s].lse1.W = F(pi); E[s].lse1.b = F(pi); E[s].lse1.beta = F(pi); E[s].lse1.gamma = F(pi);
        E[s].lse2.W = F(pi); E[s].lse2.b = F(pi); E[s].lse2.beta = F(pi); E[s].lse2.gamma = F(pi);
        E[s].mlp1W = F(pi); E[s].mlp1b = F(pi);
        E[s].mlp2W = F(pi); E[s].mlp2b = F(pi);
        E[s].p1mlp.W = F(pi); E[s].p1mlp.b = F(pi); E[s].p1mlp.beta = F(pi); E[s].p1mlp.gamma = F(pi);
        E[s].p1score = F(pi);
        E[s].p2mlp.W = F(pi); E[s].p2mlp.b = F(pi); E[s].p2mlp.beta = F(pi); E[s].p2mlp.gamma = F(pi);
        E[s].p2score = F(pi);
        E[s].sc.W = F(pi); E[s].sc.b = F(pi); E[s].sc.beta = F(pi); E[s].sc.gamma = F(pi);
    }
    MlpP fce0, fce1;
    fce0.W = F(pi); fce0.b = F(pi); fce0.beta = F(pi); fce0.gamma = F(pi);
    fce1.W = F(pi); fce1.b = F(pi); fce1.beta = F(pi); fce1.gamma = F(pi);
    const float* fce2W = F(pi); const float* fce2b = F(pi);
    const float* fcsW = F(pi);  const float* fcsb = F(pi);
    const float* midW = F(pi);  const float* midb = F(pi);

    // --- workspace arena ---------------------------------------------------
    char* ws = (char*)d_ws;
    size_t cur = 0;
    auto alloc = [&](size_t bytes) -> void* {
        void* p = ws + cur;
        cur += (bytes + 255) & ~(size_t)255;
        return p;
    };

    int*   perm   = (int*)alloc(N * sizeof(int));
    int*   invp   = (int*)alloc(N * sizeof(int));
    float* coords = (float*)alloc((size_t)B * N * 3 * sizeof(float));
    float* meanb  = (float*)alloc(1024 * sizeof(float));
    float* rstdb  = (float*)alloc(1024 * sizeof(float));
    const int DIN[4]  = {8, 32, 128, 256};
    const int DOUT[4] = {16, 64, 128, 256};
    const int SN[4]   = {8192, 2048, 512, 128};
    float* stack[4];
    for (int s = 0; s < 4; ++s)
        stack[s] = (float*)alloc((size_t)B * 2 * DOUT[s] * SN[s] * sizeof(float));
    float* xa = (float*)alloc((size_t)2 * 1024 * 1024 * sizeof(float));
    float* xb = (float*)alloc((size_t)2 * 1024 * 1024 * sizeof(float));
    size_t scratch_base = cur;

    auto L1D = [&](size_t tot) -> dim3 { return dim3((unsigned)((tot + 255) / 256)); };
    auto gemm = [&](const float* W, const float* bias, const float* X, float* Y,
                    int Cout, int Cin, int M, int act, float slope) {
        dim3 g((M + 127) / 128, (Cout + 31) / 32, B);
        k_gemm<<<g, 128, 0, stream>>>(W, bias, X, Y, Cout, Cin, M, act, slope);
    };
    auto bnorm = [&](float* Y, const MlpP& p, int C, size_t M, const float* res,
                     float* O, int act, float slope) {
        k_bn_stats<<<C, 256, 0, stream>>>(Y, B, C, (int)M, meanb, rstdb);
        k_bn_apply<<<L1D((size_t)B * C * M), 256, 0, stream>>>(
            Y, meanb, rstdb, p.gamma, p.beta, res, O, B, C, M, act, slope);
    };

    // --- fc_start + bn_start + permutation --------------------------------
    k_perm_init<<<L1D(N), 256, 0, stream>>>(perm, invp, N);
    k_perm_coords<<<L1D((size_t)B * N), 256, 0, stream>>>(points, perm, coords, B, N);

    cur = scratch_base;
    float* tmp = (float*)alloc((size_t)B * 8 * N * sizeof(float));
    k_fc_start<<<L1D((size_t)B * N), 256, 0, stream>>>(points, features, fcsW, fcsb, tmp, B, N);
    MlpP bn0; bn0.beta = bn_start_beta; bn0.gamma = bn_start_gamma;
    bnorm(tmp, bn0, 8, (size_t)N, nullptr, tmp, 2, 0.2f);
    // apply permutation to activation columns
    k_gather<<<L1D((size_t)B * 8 * N), 256, 0, stream>>>(tmp, perm, 0, xa, B, 8, N, N);

    float* x = xa;
    float* xo = xb;
    int xC = 8;
    int n = N;

    // --- encoder stages ----------------------------------------------------
    for (int s = 0; s < 4; ++s) {
        int din = DIN[s], dout = DOUT[s], h = dout / 2;
        size_t nK = (size_t)n * K;
        cur = scratch_base;
        int*   idx   = (int*)alloc((size_t)B * nK * sizeof(int));
        float* dist  = (float*)alloc((size_t)B * nK * sizeof(float));
        float* x1    = (float*)alloc((size_t)B * h * n * sizeof(float));
        float* lse10 = (float*)alloc((size_t)B * 10 * nK * sizeof(float));
        float* encb  = (float*)alloc((size_t)B * h * nK * sizeof(float));
        float* cat   = (float*)alloc((size_t)B * dout * nK * sizeof(float));
        float* score = (float*)alloc((size_t)B * dout * nK * sizeof(float));
        float* fpool = (float*)alloc((size_t)B * dout * n * sizeof(float));
        float* pm    = (float*)alloc((size_t)B * dout * n * sizeof(float));
        float* y1    = (float*)alloc((size_t)B * 2 * dout * n * sizeof(float));
        float* sc    = (float*)alloc((size_t)B * 2 * dout * n * sizeof(float));

        { dim3 g((n + 127) / 128, B); k_knn<<<g, 128, 0, stream>>>(coords, n, N * 3, idx, dist); }

        // mlp1: din -> h, leaky 0.2
        gemm(E[s].mlp1W, E[s].mlp1b, x, x1, h, din, n, 2, 0.2f);

        // lse1 + pool1
        k_lse_build<<<L1D((size_t)B * nK), 256, 0, stream>>>(coords, N * 3, idx, dist, lse10, B, n, K);
        gemm(E[s].lse1.W, E[s].lse1.b, lse10, encb, h, 10, (int)nK, 0, 0.f);
        bnorm(encb, E[s].lse1, h, nK, nullptr, encb, 1, 0.f);
        k_cat_enc_feat<<<L1D((size_t)B * h * nK), 256, 0, stream>>>(encb, x1, cat, B, h, n, K);
        gemm(E[s].p1score, nullptr, cat, score, dout, dout, (int)nK, 0, 0.f);
        k_attpool<<<L1D((size_t)B * dout * n), 256, 0, stream>>>(score, cat, fpool, B, dout, n, K);
        gemm(E[s].p1mlp.W, E[s].p1mlp.b, fpool, pm, h, dout, n, 0, 0.f);
        bnorm(pm, E[s].p1mlp, h, (size_t)n, nullptr, pm, 1, 0.f);

        // lse2 + pool2 (feat = pool1 output)
        gemm(E[s].lse2.W, E[s].lse2.b, lse10, encb, h, 10, (int)nK, 0, 0.f);
        bnorm(encb, E[s].lse2, h, nK, nullptr, encb, 1, 0.f);
        k_cat_enc_feat<<<L1D((size_t)B * h * nK), 256, 0, stream>>>(encb, pm, cat, B, h, n, K);
        gemm(E[s].p2score, nullptr, cat, score, dout, dout, (int)nK, 0, 0.f);
        k_attpool<<<L1D((size_t)B * dout * n), 256, 0, stream>>>(score, cat, fpool, B, dout, n, K);
        gemm(E[s].p2mlp.W, E[s].p2mlp.b, fpool, pm, dout, dout, n, 0, 0.f);
        bnorm(pm, E[s].p2mlp, dout, (size_t)n, nullptr, pm, 1, 0.f);

        // mlp2 + shortcut residual, leaky 0.01
        gemm(E[s].mlp2W, E[s].mlp2b, pm, y1, 2 * dout, dout, n, 0, 0.f);
        gemm(E[s].sc.W, E[s].sc.b, x, sc, 2 * dout, din, n, 0, 0.f);
        bnorm(sc, E[s].sc, 2 * dout, (size_t)n, y1, stack[s], 2, 0.01f);

        // decimate
        int n2 = n / 4;
        k_slice<<<L1D((size_t)B * 2 * dout * n2), 256, 0, stream>>>(stack[s], xo, B, 2 * dout, n2, n);
        { float* t = x; x = xo; xo = t; }
        xC = 2 * dout;
        n = n2;
    }

    // --- mid: 512 -> 512, relu (n == 32) -----------------------------------
    gemm(midW, midb, x, xo, 512, 512, n, 1, 0.f);
    { float* t = x; x = xo; xo = t; }
    xC = 512;

    // --- decoder ------------------------------------------------------------
    const int DCI[4] = {1024, 512, 256, 64};
    const int DCO[4] = {256, 128, 32, 8};
    for (int i = 0; i < 4; ++i) {
        int n_down = n;
        int n_up = 4 * n;
        int sidx = 3 - i;
        int stC = 2 * DOUT[sidx];
        cur = scratch_base;
        int*   idx1 = (int*)alloc((size_t)B * n_up * sizeof(int));
        float* xn   = (float*)alloc((size_t)B * xC * n_up * sizeof(float));
        float* cc   = (float*)alloc((size_t)B * DCI[i] * n_up * sizeof(float));

        { dim3 g((n_up + 255) / 256, B); k_nn1<<<g, 256, 0, stream>>>(coords, N * 3, n_up, n_down, idx1); }
        k_gather<<<L1D((size_t)B * xC * n_up), 256, 0, stream>>>(x, idx1, n_up, xn, B, xC, n_up, n_down);
        k_cat2<<<L1D((size_t)B * DCI[i] * n_up), 256, 0, stream>>>(xn, xC, stack[sidx], stC, cc, B, (size_t)n_up);
        gemm(dec[i].W, dec[i].b, cc, xo, DCO[i], DCI[i], n_up, 0, 0.f);
        bnorm(xo, dec[i], DCO[i], (size_t)n_up, nullptr, xo, 1, 0.f);
        { float* t = x; x = xo; xo = t; }
        xC = DCO[i];
        n = n_up;
    }

    // --- un-permute ---------------------------------------------------------
    k_gather<<<L1D((size_t)B * xC * N), 256, 0, stream>>>(x, invp, 0, xo, B, xC, N, N);
    { float* t = x; x = xo; xo = t; }

    // --- fc_end -------------------------------------------------------------
    cur = scratch_base;
    float* f64b = (float*)alloc((size_t)B * 64 * N * sizeof(float));
    float* f32b = (float*)alloc((size_t)B * 32 * N * sizeof(float));
    float* f13b = (float*)alloc((size_t)B * 16 * N * sizeof(float));
    gemm(fce0.W, fce0.b, x, f64b, 64, 8, N, 0, 0.f);
    bnorm(f64b, fce0, 64, (size_t)N, nullptr, f64b, 1, 0.f);
    gemm(fce1.W, fce1.b, f64b, f32b, 32, 64, N, 0, 0.f);
    bnorm(f32b, fce1, 32, (size_t)N, nullptr, f32b, 1, 0.f);
    gemm(fce2W, fce2b, f32b, f13b, 13, 32, N, 0, 0.f);

    k_out_transpose<<<L1D((size_t)B * 13 * N), 256, 0, stream>>>(f13b, (float*)d_out, B, 13, N);
}
